// Attention_42537356099762
// MI455X (gfx1250) — matrix operand order repaired and checked by reading, not compiled, not run on hardware
//
#include <hip/hip_runtime.h>
#include <stdint.h>

#define B_  4
#define L_  2048
#define D_  1024
#define H_  16
#define DH_ 64
#define DV_ 64
#define HD_ 1024          // H_*DH_
#define ML_ (B_*L_)       // 8192 rows

typedef __attribute__((ext_vector_type(16))) __bf16          v16bf;
typedef __attribute__((ext_vector_type(8)))  float           v8f;
typedef __attribute__((ext_vector_type(8)))  unsigned short  ushort8;
typedef int i4v __attribute__((vector_size(16)));   // matches builtin's V4i

union BFrag { v16bf v; ushort8 u2[2]; unsigned short s[16]; };

// ---- CDNA5 async global->LDS path (ASYNCcnt), feature-probed ----
#if defined(__has_builtin)
#if __has_builtin(__builtin_amdgcn_global_load_async_to_lds_b128) && \
    __has_builtin(__builtin_amdgcn_s_wait_asynccnt)
#define USE_ASYNC_LDS 1
#endif
#endif
#ifndef USE_ASYNC_LDS
#define USE_ASYNC_LDS 0
#endif

__device__ __forceinline__ void cp16_to_lds(unsigned short* dst,
                                            const unsigned short* src) {
#if USE_ASYNC_LDS
  __builtin_amdgcn_global_load_async_to_lds_b128(
      (__attribute__((address_space(1))) i4v*)(unsigned short*)src,
      (__attribute__((address_space(3))) i4v*)dst, 0, 0);
#else
  *(ushort8*)dst = *(const ushort8*)src;   // global_load_b128 + ds_store_b128
#endif
}

// Allow N outstanding async copies (the next tile) while requiring the
// current tile to have landed (ASYNCcnt completes in order).
template <int N>
__device__ __forceinline__ void wait_stage(bool more_in_flight) {
#if USE_ASYNC_LDS
  if (more_in_flight) __builtin_amdgcn_s_wait_asynccnt(N);
  else                __builtin_amdgcn_s_wait_asynccnt(0);
#endif
}

__device__ __forceinline__ unsigned short f2bf(float f) {
  union { float f; unsigned u; } cv; cv.f = f;
  unsigned u = cv.u + (0x7FFFu + ((cv.u >> 16) & 1u));   // round-to-nearest-even
  return (unsigned short)(u >> 16);
}

__device__ __forceinline__ v8f wmma_bf16(const BFrag& a, const BFrag& b, v8f c) {
  return __builtin_amdgcn_wmma_f32_16x16x32_bf16(false, a.v, false, b.v,
                                                 (short)0, c, false, false);
}

__device__ __forceinline__ void ld_frag16(BFrag& f, const unsigned short* p0,
                                          const unsigned short* p1) {
  f.u2[0] = *(const ushort8*)p0;
  f.u2[1] = *(const ushort8*)p1;
}

// ---------------- conversion kernels ----------------
__global__ void cvt_f32_bf16(const float* __restrict__ src,
                             unsigned short* __restrict__ dst, int n) {
  for (int i = blockIdx.x * blockDim.x + threadIdx.x; i < n;
       i += gridDim.x * blockDim.x)
    dst[i] = f2bf(src[i]);
}

// src: [D_][HD_] (in,out) row-major -> dst: [HD_][D_] (out,in) bf16
__global__ void cvt_transpose_w(const float* __restrict__ src,
                                unsigned short* __restrict__ dst) {
  int i = blockIdx.x * blockDim.x + threadIdx.x;   // exactly D_*HD_ threads
  int k = i >> 10, n = i & 1023;
  dst[n * D_ + k] = f2bf(src[i]);
}

// ---------------- LDS-staged GEMM machinery ----------------
// Block = 4 waves computes 64x64 tile; K-step 64. A tile (64 rows x 64 K) and
// B tile (64 cols x 64 K, from pre-transposed weights) staged to LDS with
// async copies (8 x b128 per thread per step), double buffered.
#define GT_LD 72   // 64 + 8 bf16 row pad

__device__ __forceinline__ void stage_gemm(unsigned short* at,
                                           unsigned short* bt,
                                           const unsigned short* Ag,
                                           const unsigned short* Bg,
                                           int kk, int tid) {
#pragma unroll
  for (int it = 0; it < 4; ++it) {            // A tile: 64 rows x 64 bf16
    int c = tid + it * 128;
    int row = c >> 3, col = (c & 7) * 8;
    cp16_to_lds(at + row * GT_LD + col, Ag + (size_t)row * D_ + kk + col);
  }
#pragma unroll
  for (int it = 0; it < 4; ++it) {            // B tile: 64 rows x 64 bf16
    int c = tid + it * 128;
    int row = c >> 3, col = (c & 7) * 8;
    cp16_to_lds(bt + row * GT_LD + col, Bg + (size_t)row * D_ + kk + col);
  }
}

// Inner compute: batch-load all fragments (one ds clause), then 8 WMMAs.
__device__ __forceinline__ void gemm_step(v8f* acc, const unsigned short* at,
                                          const unsigned short* bt,
                                          int g, int qi) {
  BFrag a[2], bf[2][4];
#pragma unroll
  for (int kh = 0; kh < 2; ++kh)
    ld_frag16(a[kh], at + kh * 32 + g * 8, at + kh * 32 + 16 + g * 8);
#pragma unroll
  for (int kh = 0; kh < 2; ++kh)
#pragma unroll
    for (int s = 0; s < 4; ++s) {
      const unsigned short* br =
          bt + (size_t)(s * 16 + qi) * GT_LD + kh * 32 + g * 8;
      ld_frag16(bf[kh][s], br, br + 16);
    }
#pragma unroll
  for (int kh = 0; kh < 2; ++kh)
#pragma unroll
    for (int s = 0; s < 4; ++s)
      acc[s] = wmma_bf16(a[kh], bf[kh][s], acc[s]);
}

// ---------------- QKV projection GEMM (async LDS pipelined) ----------------
__global__ __launch_bounds__(128) void qkv_gemm(
    const unsigned short* __restrict__ Xb,
    const unsigned short* __restrict__ WqT,
    const unsigned short* __restrict__ WkT,
    const unsigned short* __restrict__ WvT,
    unsigned short* __restrict__ Qb,
    unsigned short* __restrict__ Kb,
    unsigned short* __restrict__ Vt) {
  __shared__ __align__(16) unsigned short atile[2][64 * GT_LD];
  __shared__ __align__(16) unsigned short btile[2][64 * GT_LD];
  const int tid  = threadIdx.x;
  const int lane = tid & 31;
  const int wave = tid >> 5;
  const int g    = lane >> 4;
  const int qi   = lane & 15;
  const int z    = blockIdx.z;
  const unsigned short* Wt = (z == 0) ? WqT : (z == 1) ? WkT : WvT;
  const int mb = blockIdx.x * 64;
  const int nb = blockIdx.y * 64;
  const unsigned short* Ag = Xb + (size_t)mb * D_;
  const unsigned short* Bg = Wt + (size_t)nb * D_;
  v8f acc[4] = {v8f{}, v8f{}, v8f{}, v8f{}};
  int cur = 0;
  stage_gemm(atile[0], btile[0], Ag, Bg, 0, tid);
  for (int kk = 0; kk < D_; kk += 64) {
    const bool more = (kk + 64) < D_;
    if (more)
      stage_gemm(atile[cur ^ 1], btile[cur ^ 1], Ag, Bg, kk + 64, tid);
    wait_stage<8>(more);
    __syncthreads();
    gemm_step(acc, atile[cur] + (size_t)(wave * 16 + qi) * GT_LD, btile[cur],
              g, qi);
    __syncthreads();
    cur ^= 1;
  }
  const float scale = (z == 0) ? 0.125f : 1.0f;   // fold 1/sqrt(dh) into Q
  const int m0 = mb + wave * 16;
#pragma unroll
  for (int s = 0; s < 4; ++s) {
#pragma unroll
    for (int r = 0; r < 8; ++r) {
      int M = m0 + r + g * 8;
      int N = nb + s * 16 + qi;
      int b = M >> 11, l = M & (L_ - 1);
      int h = N >> 6,  dd = N & 63;
      unsigned short v = f2bf(acc[s][r] * scale);
      if (z == 0)      Qb[((size_t)((b * H_ + h) * L_ + l)) * DH_ + dd] = v;
      else if (z == 1) Kb[((size_t)((b * H_ + h) * L_ + l)) * DH_ + dd] = v;
      else             Vt[((size_t)((b * H_ + h) * DV_ + dd)) * L_ + l] = v;
    }
  }
}

// ---------------- causal flash attention ----------------
// Block = 4 waves = 64 queries of one (b,h). Each 32-key step, the block
// cooperatively stages K (32x64) and V^T (64x32) into double-buffered LDS
// via async global->LDS, then each wave computes transposed flash attention:
// S^T = K x Q^T, online softmax (per-lane stats, one half-wave shuffle),
// O^T += V^T x P^T.
#define KT_LD 72   // 64 + 8 bf16 pad
#define VT_LD 40   // 32 + 8 bf16 pad

__device__ __forceinline__ void stage_tiles(unsigned short* kt,
                                            unsigned short* vt,
                                            const unsigned short* Kg,
                                            const unsigned short* Vg,
                                            int kb, int tid) {
#pragma unroll
  for (int it = 0; it < 2; ++it) {            // K tile: 32 rows x 64 bf16
    int c = tid + it * 128;
    int row = c >> 3, col = (c & 7) * 8;
    cp16_to_lds(kt + row * KT_LD + col, Kg + (size_t)(kb + row) * DH_ + col);
  }
#pragma unroll
  for (int it = 0; it < 2; ++it) {            // V^T tile: 64 rows x 32 bf16
    int c = tid + it * 128;
    int row = c >> 2, col = (c & 3) * 8;
    cp16_to_lds(vt + row * VT_LD + col, Vg + (size_t)row * L_ + kb + col);
  }
}

__global__ __launch_bounds__(128) void flash_attn(
    const unsigned short* __restrict__ Qb,
    const unsigned short* __restrict__ Kb,
    const unsigned short* __restrict__ Vt,
    unsigned short* __restrict__ Ctx) {
  __shared__ __align__(16) unsigned short ktile[2][32 * KT_LD];
  __shared__ __align__(16) unsigned short vtile[2][64 * VT_LD];
  const int lane = threadIdx.x & 31;
  const int wave = threadIdx.x >> 5;
  const int g    = lane >> 4;
  const int qi   = lane & 15;
  const int h = blockIdx.y, b = blockIdx.z;
  const int qb = blockIdx.x * 64;
  const int q0 = qb + wave * 16;
  const size_t headQK = (size_t)(b * H_ + h) * L_ * DH_;
  const size_t headV  = (size_t)(b * H_ + h) * DV_ * L_;
  const unsigned short* Kg = Kb + headQK;
  const unsigned short* Vg = Vt + headV;

  // Q^T B-fragments (dh 0..31, 32..63) — loop invariant, Q pre-scaled
  BFrag bq0, bq1;
  {
    const unsigned short* qrow = Qb + headQK + (size_t)(q0 + qi) * DH_;
    ld_frag16(bq0, qrow + g * 8, qrow + 16 + g * 8);
    ld_frag16(bq1, qrow + 32 + g * 8, qrow + 48 + g * 8);
  }
  v8f o0 = {}, o1 = {}, o2 = {}, o3 = {};
  float mrow = -__builtin_inff(), lrow = 0.0f;
  const int kend_w   = q0 + 16;        // this wave's causal key bound
  const int kend_blk = qb + 64;        // block's key bound (last wave)

  int cur = 0;
  stage_tiles(ktile[0], vtile[0], Kg, Vg, 0, threadIdx.x);
  for (int kb = 0; kb < kend_blk; kb += 32) {
    const bool more = (kb + 32) < kend_blk;
    if (more)
      stage_tiles(ktile[cur ^ 1], vtile[cur ^ 1], Kg, Vg, kb + 32, threadIdx.x);
    wait_stage<4>(more);               // own tile landed; next tile in flight
    __syncthreads();
    if (kb < kend_w) {                 // wave-uniform predicate (EXEC stays full)
      const unsigned short* ktb = ktile[cur];
      const unsigned short* vtb = vtile[cur];
      // batch-load the 4 K A-fragments, then 4 QK^T WMMAs
      BFrag ak[4];
      {
        const unsigned short* kr0 = ktb + (size_t)qi * KT_LD;
        const unsigned short* kr1 = ktb + (size_t)(16 + qi) * KT_LD;
        ld_frag16(ak[0], kr0 + g * 8,      kr0 + 16 + g * 8);
        ld_frag16(ak[1], kr0 + 32 + g * 8, kr0 + 48 + g * 8);
        ld_frag16(ak[2], kr1 + g * 8,      kr1 + 16 + g * 8);
        ld_frag16(ak[3], kr1 + 32 + g * 8, kr1 + 48 + g * 8);
      }
      v8f st0 = {}, st1 = {};
      st0 = wmma_bf16(ak[0], bq0, st0);
      st0 = wmma_bf16(ak[1], bq1, st0);
      st1 = wmma_bf16(ak[2], bq0, st1);
      st1 = wmma_bf16(ak[3], bq1, st1);
      const int q = q0 + qi;
#pragma unroll
      for (int r = 0; r < 8; ++r) {    // causal mask from coordinates
        int key = kb + r + g * 8;
        if (key > q)      st0[r] = -1e30f;
        if (key + 16 > q) st1[r] = -1e30f;
      }
      float mk = fmaxf(st0[0], st1[0]);
#pragma unroll
      for (int r = 1; r < 8; ++r) mk = fmaxf(mk, fmaxf(st0[r], st1[r]));
      mk = fmaxf(mk, __shfl_xor(mk, 16));
      float mnew = fmaxf(mrow, mk);
      float corr = __expf(mrow - mnew);
#pragma unroll
      for (int r = 0; r < 8; ++r) {
        st0[r] = __expf(st0[r] - mnew);
        st1[r] = __expf(st1[r] - mnew);
      }
      float ps = 0.f;
#pragma unroll
      for (int r = 0; r < 8; ++r) ps += st0[r] + st1[r];
      ps += __shfl_xor(ps, 16);
      lrow = lrow * corr + ps;
      mrow = mnew;
#pragma unroll
      for (int r = 0; r < 8; ++r) {
        o0[r] *= corr; o1[r] *= corr; o2[r] *= corr; o3[r] *= corr;
      }
      // pack P^T into B-fragment layout via half-wave exchange
      float ax[8], bx[8];
#pragma unroll
      for (int r = 0; r < 8; ++r) {
        ax[r] = __shfl_xor(st0[r], 16);
        bx[r] = __shfl_xor(st1[r], 16);
      }
      BFrag pf;
#pragma unroll
      for (int e = 0; e < 8; ++e) pf.s[e]     = f2bf(g ? bx[e]  : st0[e]);
#pragma unroll
      for (int e = 0; e < 8; ++e) pf.s[e + 8] = f2bf(g ? st1[e] : ax[e]);
      // batch-load the 4 V^T A-fragments, then 4 PV WMMAs
      BFrag av[4];
#pragma unroll
      for (int s = 0; s < 4; ++s) {
        const unsigned short* vr = vtb + (size_t)(s * 16 + qi) * VT_LD;
        ld_frag16(av[s], vr + g * 16, vr + g * 16 + 8);
      }
      o0 = wmma_bf16(av[0], pf, o0);
      o1 = wmma_bf16(av[1], pf, o1);
      o2 = wmma_bf16(av[2], pf, o2);
      o3 = wmma_bf16(av[3], pf, o3);
    }
    __syncthreads();
    cur ^= 1;
  }
  const float inv = 1.0f / lrow;
  const size_t crow = ((size_t)(b * L_ + q0 + qi)) * HD_ + h * DV_;
  v8f os[4] = {o0, o1, o2, o3};
#pragma unroll
  for (int s = 0; s < 4; ++s)
#pragma unroll
    for (int r = 0; r < 8; ++r) {
      int dv = s * 16 + r + g * 8;
      Ctx[crow + dv] = f2bf(os[s][r] * inv);
    }
}

// ---------------- output projection: ctx @ Wo + bo -> fp32 ----------------
__global__ __launch_bounds__(128) void out_gemm(
    const unsigned short* __restrict__ Ctx,
    const unsigned short* __restrict__ WoT,
    const float* __restrict__ bo,
    float* __restrict__ Out) {
  __shared__ __align__(16) unsigned short atile[2][64 * GT_LD];
  __shared__ __align__(16) unsigned short btile[2][64 * GT_LD];
  const int tid  = threadIdx.x;
  const int lane = tid & 31;
  const int wave = tid >> 5;
  const int g    = lane >> 4;
  const int qi   = lane & 15;
  const int mb = blockIdx.x * 64;
  const int nb = blockIdx.y * 64;
  const unsigned short* Ag = Ctx + (size_t)mb * HD_;
  const unsigned short* Bg = WoT + (size_t)nb * HD_;
  v8f acc[4] = {v8f{}, v8f{}, v8f{}, v8f{}};
  int cur = 0;
  stage_gemm(atile[0], btile[0], Ag, Bg, 0, tid);
  for (int kk = 0; kk < HD_; kk += 64) {
    const bool more = (kk + 64) < HD_;
    if (more)
      stage_gemm(atile[cur ^ 1], btile[cur ^ 1], Ag, Bg, kk + 64, tid);
    wait_stage<8>(more);
    __syncthreads();
    gemm_step(acc, atile[cur] + (size_t)(wave * 16 + qi) * GT_LD, btile[cur],
              g, qi);
    __syncthreads();
    cur ^= 1;
  }
  const int m0 = mb + wave * 16;
#pragma unroll
  for (int s = 0; s < 4; ++s)
#pragma unroll
    for (int r = 0; r < 8; ++r) {
      int M = m0 + r + g * 8;
      int N = nb + s * 16 + qi;
      Out[(size_t)M * D_ + N] = acc[s][r] + bo[N];
    }
}

// ---------------- launcher ----------------
extern "C" void kernel_launch(void* const* d_in, const int* in_sizes, int n_in,
                              void* d_out, int out_size, void* d_ws, size_t ws_size,
                              hipStream_t stream) {
  const float* x  = (const float*)d_in[0];
  const float* Wq = (const float*)d_in[1];
  const float* Wk = (const float*)d_in[2];
  const float* Wv = (const float*)d_in[3];
  const float* Wo = (const float*)d_in[4];
  const float* bo = (const float*)d_in[5];
  // d_in[6] = causal mask: recomputed analytically in-kernel
  float* out = (float*)d_out;

  char* ws = (char*)d_ws;                    // ~88 MB used, 1 MB-aligned slabs
  unsigned short* Xb  = (unsigned short*)(ws + 0);           // 16 MB
  unsigned short* WqT = (unsigned short*)(ws + 16777216);    //  2 MB
  unsigned short* WkT = (unsigned short*)(ws + 18874368);    //  2 MB
  unsigned short* WvT = (unsigned short*)(ws + 20971520);    //  2 MB
  unsigned short* WoT = (unsigned short*)(ws + 23068672);    //  2 MB
  unsigned short* Qb  = (unsigned short*)(ws + 25165824);    // 16 MB
  unsigned short* Kb  = (unsigned short*)(ws + 41943040);    // 16 MB
  unsigned short* Vt  = (unsigned short*)(ws + 58720256);    // 16 MB
  unsigned short* Ctx = (unsigned short*)(ws + 75497472);    // 16 MB

  cvt_f32_bf16<<<4096, 256, 0, stream>>>(x, Xb, ML_ * D_);
  cvt_transpose_w<<<4096, 256, 0, stream>>>(Wq, WqT);
  cvt_transpose_w<<<4096, 256, 0, stream>>>(Wk, WkT);
  cvt_transpose_w<<<4096, 256, 0, stream>>>(Wv, WvT);
  cvt_transpose_w<<<4096, 256, 0, stream>>>(Wo, WoT);
  qkv_gemm<<<dim3(ML_ / 64, HD_ / 64, 3), 128, 0, stream>>>(
      Xb, WqT, WkT, WvT, Qb, Kb, Vt);
  flash_attn<<<dim3(L_ / 64, H_, B_), 128, 0, stream>>>(Qb, Kb, Vt, Ctx);
  out_gemm<<<dim3(ML_ / 64, D_ / 64), 128, 0, stream>>>(Ctx, WoT, bo, out);
}
